// Router_24764781428916
// MI455X (gfx1250) — compile-verified, matrix operand order checked by reading, not run on hardware
//
#include <hip/hip_runtime.h>
#include <hip/hip_bf16.h>

typedef __attribute__((ext_vector_type(16))) __bf16 v16bf;
typedef __attribute__((ext_vector_type(8)))  float  v8f;

#define N_TOK 16384
#define DIM   4096
#define NEXP  64
#define TOPK  8
#define NCHUNK (DIM / 32)          // 128 K-chunks of 32
#define SWZ_PER_CHUNK 4096         // bf16 elems per chunk: 2 sel * 4 tile * 32 lane * 16
#define SWZ_TOTAL (NCHUNK * SWZ_PER_CHUNK)   // 524288 bf16 = 1 MB

#define WMMA_BF16(A, B, C) \
    __builtin_amdgcn_wmma_f32_16x16x32_bf16(false, (A), false, (B), (short)0, (C), false, false)

// ---------------------------------------------------------------------------
// Kernel 0: zero the stats accumulators (importance[64], load[64])
// ---------------------------------------------------------------------------
__global__ void router_zero_stats(float* stats) {
    stats[threadIdx.x] = 0.0f;   // launched with 128 threads
}

// ---------------------------------------------------------------------------
// Kernel 1: pre-swizzle W into the WMMA B-fragment register layout, split
// into bf16 hi/lo.  Layout: [kc][sel(hi/lo)][tile][lane][j0..15] bf16.
// Fragment K split per ISA: elems 0..7 = K{koff..koff+7}, 8..15 = K{16+koff..}
// with koff = (lane<16) ? 0 : 8;  column = tile*16 + (lane&15).
// ---------------------------------------------------------------------------
__global__ __launch_bounds__(256) void router_pack_w(
    const float* __restrict__ W, __bf16* __restrict__ Wswz)
{
    const int idx  = blockIdx.x * 256 + threadIdx.x;   // [0, SWZ_TOTAL)
    const int j    = idx & 15;
    const int lane = (idx >> 4) & 31;
    const int tile = (idx >> 9) & 3;
    const int sel  = (idx >> 11) & 1;
    const int kc   = idx >> 12;

    const int e    = tile * 16 + (lane & 15);
    const int koff = (lane < 16) ? 0 : 8;
    const int kk   = (j < 8) ? (koff + j) : (16 + koff + (j - 8));
    const int k    = kc * 32 + kk;

    const float  w  = W[(size_t)k * NEXP + e];
    const __bf16 hi = (__bf16)w;
    Wswz[idx] = sel ? (__bf16)(w - (float)hi) : hi;
}

// ---------------------------------------------------------------------------
// Kernel 2: logits = x @ W + noise via split-bf16 WMMA (hi*hi + hi*lo + lo*hi)
// Block: 256 threads (8 waves). Each wave owns 16 unique rows x all 64
// experts (4 accumulators); block covers 128 rows. B fragments come from LDS
// pre-swizzled, so a fragment is one aligned 32-byte vector load.
// ---------------------------------------------------------------------------
__global__ __launch_bounds__(256) void router_gemm(
    const float* __restrict__ x, const __bf16* __restrict__ Wswz,
    const float* __restrict__ noise, float* __restrict__ logits)
{
    __shared__ __align__(32) __bf16 sW[2][2][4][32][16]; // [buf][sel][tile][lane][j]

    const int tid  = threadIdx.x;
    const int wave = tid >> 5;
    const int lane = tid & 31;

    const int rowBase = blockIdx.x * 128 + wave * 16;
    const int aRow    = rowBase + (lane & 15);
    const int koff    = (lane < 16) ? 0 : 8;
    const float* __restrict__ xrow = x + (size_t)aRow * DIM;

    uint4*       sWv = (uint4*)&sW[0][0][0][0][0];   // 512 uint4 per buffer
    const uint4* Wv  = (const uint4*)Wswz;           // 512 uint4 per chunk

    v8f acc0 = {}, acc1 = {}, acc2 = {}, acc3 = {};

    // --- stage chunk 0 (coalesced: 2x b128 per thread) ---
    sWv[2 * tid]     = Wv[2 * tid];
    sWv[2 * tid + 1] = Wv[2 * tid + 1];
    __syncthreads();

    for (int kc = 0; kc < NCHUNK; ++kc) {
        const int buf = kc & 1;
        const int k0  = kc * 32;

        // prefetch x ahead (global_prefetch_b8)
        __builtin_prefetch(xrow + k0 + 128, 0, 0);

        // --- stage next chunk into the other LDS buffer ---
        if (kc + 1 < NCHUNK) {
            const uint4* src = Wv + (size_t)(kc + 1) * 512;
            uint4*       dst = sWv + (buf ^ 1) * 512;
            dst[2 * tid]     = src[2 * tid];
            dst[2 * tid + 1] = src[2 * tid + 1];
        }

        // --- A fragment (loaded once per x element): split bf16 hi/lo ---
        v16bf a_hi, a_lo;
        #pragma unroll
        for (int j = 0; j < 8; ++j) {
            const float  f0 = xrow[k0 + koff + j];
            const float  f1 = xrow[k0 + 16 + koff + j];
            const __bf16 h0 = (__bf16)f0;
            const __bf16 h1 = (__bf16)f1;
            a_hi[j]     = h0;  a_lo[j]     = (__bf16)(f0 - (float)h0);
            a_hi[8 + j] = h1;  a_lo[8 + j] = (__bf16)(f1 - (float)h1);
        }

        // --- 4 expert tiles: 8 vector LDS reads + 12 WMMAs per chunk ---
        {
            const v16bf bh = *(const v16bf*)&sW[buf][0][0][lane][0];
            const v16bf bl = *(const v16bf*)&sW[buf][1][0][lane][0];
            acc0 = WMMA_BF16(a_hi, bh, acc0);
            acc0 = WMMA_BF16(a_hi, bl, acc0);
            acc0 = WMMA_BF16(a_lo, bh, acc0);
        }
        {
            const v16bf bh = *(const v16bf*)&sW[buf][0][1][lane][0];
            const v16bf bl = *(const v16bf*)&sW[buf][1][1][lane][0];
            acc1 = WMMA_BF16(a_hi, bh, acc1);
            acc1 = WMMA_BF16(a_hi, bl, acc1);
            acc1 = WMMA_BF16(a_lo, bh, acc1);
        }
        {
            const v16bf bh = *(const v16bf*)&sW[buf][0][2][lane][0];
            const v16bf bl = *(const v16bf*)&sW[buf][1][2][lane][0];
            acc2 = WMMA_BF16(a_hi, bh, acc2);
            acc2 = WMMA_BF16(a_hi, bl, acc2);
            acc2 = WMMA_BF16(a_lo, bh, acc2);
        }
        {
            const v16bf bh = *(const v16bf*)&sW[buf][0][3][lane][0];
            const v16bf bl = *(const v16bf*)&sW[buf][1][3][lane][0];
            acc3 = WMMA_BF16(a_hi, bh, acc3);
            acc3 = WMMA_BF16(a_hi, bl, acc3);
            acc3 = WMMA_BF16(a_lo, bh, acc3);
        }

        __syncthreads();
    }

    // --- epilogue: C/D layout -> VGPR v = row (v or v+8), col = lane&15 ---
    const int rOff = (lane < 16) ? 0 : 8;
    #pragma unroll
    for (int v = 0; v < 8; ++v) {
        const int    r = rowBase + rOff + v;
        const size_t o = (size_t)r * NEXP + (lane & 15);
        logits[o +  0] = acc0[v] + noise[o +  0];
        logits[o + 16] = acc1[v] + noise[o + 16];
        logits[o + 32] = acc2[v] + noise[o + 32];
        logits[o + 48] = acc3[v] + noise[o + 48];
    }
}

// ---------------------------------------------------------------------------
// Kernel 3: per-row softmax + top-8 + normalize; accumulate importance/load.
// One wave32 per row (2 experts per lane). 8 waves per block.
// ---------------------------------------------------------------------------
__global__ __launch_bounds__(256) void router_topk(
    const float* __restrict__ logits, float* __restrict__ gates_out,
    float* __restrict__ idx_out, float* __restrict__ g_imp,
    float* __restrict__ g_load)
{
    __shared__ float s_imp[NEXP];
    __shared__ float s_load[NEXP];

    const int tid  = threadIdx.x;
    const int wave = tid >> 5;
    const int lane = tid & 31;

    if (tid < NEXP) { s_imp[tid] = 0.0f; s_load[tid] = 0.0f; }
    __syncthreads();

    const int row = blockIdx.x * 8 + wave;
    const size_t base = (size_t)row * NEXP;

    float g0 = logits[base + lane];
    float g1 = logits[base + 32 + lane];

    // softmax: butterfly max + sum across wave32
    float m = fmaxf(g0, g1);
    #pragma unroll
    for (int off = 16; off > 0; off >>= 1)
        m = fmaxf(m, __shfl_xor(m, off, 32));
    float e0 = __expf(g0 - m);
    float e1 = __expf(g1 - m);
    float s = e0 + e1;
    #pragma unroll
    for (int off = 16; off > 0; off >>= 1)
        s += __shfl_xor(s, off, 32);
    const float inv = 1.0f / s;
    const float p0 = e0 * inv;
    const float p1 = e1 * inv;

    atomicAdd(&s_imp[lane],      p0);
    atomicAdd(&s_imp[32 + lane], p1);

    // top-8 by repeated argmax on (float_bits<<32 | expert) keys
    float c0 = p0, c1 = p1;
    float ssum = 0.0f;
    float myv = 0.0f;
    int   myi = 0;
    #pragma unroll
    for (int t = 0; t < TOPK; ++t) {
        unsigned k0b = __float_as_uint(c0);   // gates > 0 -> bit order == value order
        unsigned k1b = __float_as_uint(c1);
        unsigned long long key0 = ((unsigned long long)k0b << 32) | (unsigned)lane;
        unsigned long long key1 = ((unsigned long long)k1b << 32) | (unsigned)(32 + lane);
        unsigned long long best = (key0 > key1) ? key0 : key1;
        #pragma unroll
        for (int off = 16; off > 0; off >>= 1) {
            unsigned hi = (unsigned)__shfl_xor((int)(best >> 32), off, 32);
            unsigned lo = (unsigned)__shfl_xor((int)(best & 0xffffffffu), off, 32);
            unsigned long long other = ((unsigned long long)hi << 32) | lo;
            if (other > best) best = other;
        }
        const int   bi = (int)(best & 0xffffffffu);
        const float bv = __uint_as_float((unsigned)(best >> 32));
        ssum += bv;
        if (lane == t) { myv = bv; myi = bi; }
        if (bi == lane)      c0 = 0.0f;       // mask out winner
        if (bi == 32 + lane) c1 = 0.0f;
    }

    const float rinv = 1.0f / ssum;
    if (lane < TOPK) {
        gates_out[(size_t)row * TOPK + lane] = myv * rinv;
        idx_out[(size_t)row * TOPK + lane]   = (float)myi;
        atomicAdd(&s_load[myi], 1.0f);
    }
    __syncthreads();

    if (tid < NEXP) {
        atomicAdd(&g_imp[tid],  s_imp[tid]);
        atomicAdd(&g_load[tid], s_load[tid]);
    }
}

// ---------------------------------------------------------------------------
// Kernel 4: loss = sum_e importance[e]/N * load[e]/(N*K) * E   (1 wave)
// ---------------------------------------------------------------------------
__global__ void router_loss(const float* __restrict__ g_imp,
                            const float* __restrict__ g_load,
                            float* __restrict__ loss_out)
{
    const int lane = threadIdx.x;             // 32 threads
    const float ni = 1.0f / (float)N_TOK;
    const float nl = 1.0f / (float)(N_TOK * TOPK);
    float v = g_imp[lane]      * ni * g_load[lane]      * nl
            + g_imp[lane + 32] * ni * g_load[lane + 32] * nl;
    #pragma unroll
    for (int off = 16; off > 0; off >>= 1)
        v += __shfl_xor(v, off, 32);
    if (lane == 0) loss_out[0] = v * (float)NEXP;
}

// ---------------------------------------------------------------------------
extern "C" void kernel_launch(void* const* d_in, const int* in_sizes, int n_in,
                              void* d_out, int out_size, void* d_ws, size_t ws_size,
                              hipStream_t stream)
{
    const float* x     = (const float*)d_in[0];
    const float* W     = (const float*)d_in[1];
    const float* noise = (const float*)d_in[2];
    // d_in[3] = top_k scalar; fixed at compile time (TOPK = 8)

    float* out       = (float*)d_out;
    float* gates_out = out;                                   // N*8
    float* idx_out   = out + (size_t)N_TOK * TOPK;            // N*8 (as float)
    float* loss_out  = out + (size_t)2 * N_TOK * TOPK;        // 1

    float*  logits = (float*)d_ws;                            // N*E f32 = 4 MB
    float*  g_imp  = logits + (size_t)N_TOK * NEXP;           // 64
    float*  g_load = g_imp + NEXP;                            // 64
    __bf16* Wswz   = (__bf16*)(g_load + NEXP);                // 1 MB swizzled W

    router_zero_stats<<<1, 128, 0, stream>>>(g_imp);          // zeros imp+load
    router_pack_w<<<SWZ_TOTAL / 256, 256, 0, stream>>>(W, Wswz);
    router_gemm<<<N_TOK / 128, 256, 0, stream>>>(x, Wswz, noise, logits);
    router_topk<<<N_TOK / 8, 256, 0, stream>>>(logits, gates_out, idx_out,
                                               g_imp, g_load);
    router_loss<<<1, 32, 0, stream>>>(g_imp, g_load, loss_out);
}
